// PretrainGNNEncoder_46480136077495
// MI455X (gfx1250) — compile-verified, matrix operand order checked by reading, not run on hardware
//
#include <hip/hip_runtime.h>
#include <hip/hip_bf16.h>

// ---------- WMMA vector types (CDNA5 / gfx1250, wave32) ----------
typedef __attribute__((ext_vector_type(16))) __bf16 v16bf;
typedef __attribute__((ext_vector_type(8)))  __bf16 v8bf;
typedef __attribute__((ext_vector_type(8)))  float  v8f;

// ======================================================================
// GEMM: out[M x 256] = A[M x K](bf16, row-major) * W[256 x K](bf16)^T
// One wave32 computes a 16x64 output tile: 4 accumulators, A reused 4x.
// A-fragment layout (16-bit A 16x32, ISA 7.12.2): lane m = lane&15,
// half = lane>>4; VGPRs 0..3 hold K = half*8 .. +7 (packed pairs),
// VGPRs 4..7 hold K = 16+half*8 .. +7  -> two contiguous 16B loads.
// B (32x16) uses the mirrored layout: lane holds column n = lane&15,
// i.e. row (c0+n) of W with the same K pattern (W^T column = W row).
// C/D: VGPR j -> row (j + half*8), col = lane&15 (per 16-col subtile).
// ======================================================================
__global__ void __launch_bounds__(128)
k_gemm_bf16(const __bf16* __restrict__ A, const __bf16* __restrict__ W,
            float* __restrict__ out, int K, int M)
{
    const int lane = threadIdx.x & 31;
    const int wave = threadIdx.x >> 5;        // 0..3 -> 64-col quad
    const int m0   = blockIdx.x * 16;         // M is a multiple of 16
    const int c0   = wave * 64;
    const int half = lane >> 4;               // 0/1 -> K-half select
    const int mr   = lane & 15;

    const __bf16* __restrict__ arow = A + (size_t)(m0 + mr) * K;
    const __bf16* __restrict__ wrow = W + (size_t)(c0 + mr) * K;
    const size_t wstep = (size_t)16 * K;      // next 16-column subtile

    v8f acc0 = {}, acc1 = {}, acc2 = {}, acc3 = {};

    for (int k0 = 0; k0 < K; k0 += 32) {
        const int kA = k0 + half * 8;
        const int kB = k0 + 16 + half * 8;

        union { v16bf v; v8bf h[2]; } ua, ub;
        ua.h[0] = *(const v8bf*)(arow + kA);
        ua.h[1] = *(const v8bf*)(arow + kB);
        const v16bf a = ua.v;

        ub.h[0] = *(const v8bf*)(wrow + kA);
        ub.h[1] = *(const v8bf*)(wrow + kB);
        acc0 = __builtin_amdgcn_wmma_f32_16x16x32_bf16(
                   false, a, false, ub.v, (short)0, acc0, false, false);

        ub.h[0] = *(const v8bf*)(wrow + wstep + kA);
        ub.h[1] = *(const v8bf*)(wrow + wstep + kB);
        acc1 = __builtin_amdgcn_wmma_f32_16x16x32_bf16(
                   false, a, false, ub.v, (short)0, acc1, false, false);

        ub.h[0] = *(const v8bf*)(wrow + 2 * wstep + kA);
        ub.h[1] = *(const v8bf*)(wrow + 2 * wstep + kB);
        acc2 = __builtin_amdgcn_wmma_f32_16x16x32_bf16(
                   false, a, false, ub.v, (short)0, acc2, false, false);

        ub.h[0] = *(const v8bf*)(wrow + 3 * wstep + kA);
        ub.h[1] = *(const v8bf*)(wrow + 3 * wstep + kB);
        acc3 = __builtin_amdgcn_wmma_f32_16x16x32_bf16(
                   false, a, false, ub.v, (short)0, acc3, false, false);
    }

    float* __restrict__ orow = out + (size_t)(m0 + half * 8) * 256 + c0 + mr;
#pragma unroll
    for (int j = 0; j < 8; ++j) {
        orow[(size_t)j * 256 +  0] = acc0[j];
        orow[(size_t)j * 256 + 16] = acc1[j];
        orow[(size_t)j * 256 + 32] = acc2[j];
        orow[(size_t)j * 256 + 48] = acc3[j];
    }
}

// ---------- convert float -> bf16 with optional K padding ----------
__global__ void k_pad_bf16(const float* __restrict__ in, __bf16* __restrict__ out,
                           int rows, int Fin, int Kp)
{
    int i = blockIdx.x * blockDim.x + threadIdx.x;
    int total = rows * Kp;
    if (i >= total) return;
    int r = i / Kp, k = i - r * Kp;
    float v = (k < Fin) ? in[(size_t)r * Fin + k] : 0.0f;
    out[i] = (__bf16)v;
}

__global__ void k_fill(float* __restrict__ p, float val, int n)
{
    int i = blockIdx.x * blockDim.x + threadIdx.x;
    if (i < n) p[i] = val;
}

__global__ void k_deg_accum(const int* __restrict__ dst, float* __restrict__ deg, int E)
{
    int e = blockIdx.x * blockDim.x + threadIdx.x;
    if (e < E) atomicAdd(deg + dst[e], 1.0f);
}

__global__ void k_rsqrt_inplace(float* __restrict__ p, int n)
{
    int i = blockIdx.x * blockDim.x + threadIdx.x;
    if (i < n) p[i] = rsqrtf(p[i]);
}

__global__ void k_edge_norm(const int* __restrict__ src, const int* __restrict__ dst,
                            const float* __restrict__ dinv, float* __restrict__ nrm, int E)
{
    int e = blockIdx.x * blockDim.x + threadIdx.x;
    if (e < E) nrm[e] = dinv[src[e]] * dinv[dst[e]];
}

// h = relu(t + b), plus bf16 copy for next GEMM
__global__ void k_bias_relu(const float* __restrict__ t, const float* __restrict__ b,
                            float* __restrict__ h, __bf16* __restrict__ hb, int total)
{
    int i = blockIdx.x * blockDim.x + threadIdx.x;
    if (i >= total) return;
    float v = fmaxf(t[i] + b[i & 255], 0.0f);
    h[i] = v;
    hb[i] = (__bf16)v;
}

// agg = conv_b[c] + m * dinv[r]^2   (self-loop term folded in, no atomics)
__global__ void k_agg_init(const float* __restrict__ m, const float* __restrict__ dinv,
                           const float* __restrict__ cb, float* __restrict__ agg, int total)
{
    int i = blockIdx.x * blockDim.x + threadIdx.x;
    if (i >= total) return;
    int r = i >> 8, c = i & 255;
    float di = dinv[r];
    agg[i] = cb[c] + m[i] * di * di;
}

// one 256-thread block per edge: agg[dst] += m[src] * norm  (f32 atomics)
__global__ void __launch_bounds__(256)
k_scatter(const int* __restrict__ src, const int* __restrict__ dst,
          const float* __restrict__ nrm, const float* __restrict__ m,
          float* __restrict__ agg, int E)
{
    int e = blockIdx.x;
    if (e >= E) return;
    int c = threadIdx.x;
    int s = src[e], d = dst[e];
    float v = m[(size_t)s * 256 + c] * nrm[e];
    atomicAdd(agg + (size_t)d * 256 + c, v);
}

// per-channel partial sums: thread t owns channel t, coalesced row reads
__global__ void __launch_bounds__(256)
k_stats_partial(const float* __restrict__ agg, float* __restrict__ sum,
                float* __restrict__ sumsq, int N)
{
    int c = threadIdx.x;
    float s = 0.0f, s2 = 0.0f;
    for (int r = blockIdx.x; r < N; r += gridDim.x) {
        float v = agg[(size_t)r * 256 + c];
        s += v; s2 += v * v;
    }
    atomicAdd(sum + c, s);
    atomicAdd(sumsq + c, s2);
}

// sum -> mu, sumsq -> rsqrt(var + eps)
__global__ void k_stats_final(float* __restrict__ sum, float* __restrict__ sumsq, int N)
{
    int c = threadIdx.x;
    float inv_n = 1.0f / (float)N;
    float mu = sum[c] * inv_n;
    float var = sumsq[c] * inv_n - mu * mu;
    sum[c] = mu;
    sumsq[c] = rsqrtf(var + 1e-5f);
}

// h_out = relu((agg - mu)*istd*gamma + beta) + h_res ; also bf16 copy
__global__ void k_bn_apply(const float* __restrict__ agg, const float* __restrict__ mu,
                           const float* __restrict__ istd, const float* __restrict__ gamma,
                           const float* __restrict__ beta, const float* __restrict__ hres,
                           float* __restrict__ hout, __bf16* __restrict__ hbout, int total)
{
    int i = blockIdx.x * blockDim.x + threadIdx.x;
    if (i >= total) return;
    int c = i & 255;
    float v = (agg[i] - mu[c]) * istd[c] * gamma[c] + beta[c];
    v = fmaxf(v, 0.0f) + hres[i];
    hout[i] = v;
    hbout[i] = (__bf16)v;
}

// ======================================================================
static inline size_t align256(size_t x) { return (x + 255) & ~(size_t)255; }

extern "C" void kernel_launch(void* const* d_in, const int* in_sizes, int n_in,
                              void* d_out, int out_size, void* d_ws, size_t ws_size,
                              hipStream_t stream)
{
    const float* x        = (const float*)d_in[0];
    const int*   ei       = (const int*)  d_in[1];
    const float* W_in     = (const float*)d_in[2];
    const float* b_in     = (const float*)d_in[3];
    const float* conv_W   = (const float*)d_in[4];
    const float* conv_b   = (const float*)d_in[5];
    const float* bn_gamma = (const float*)d_in[6];
    const float* bn_beta  = (const float*)d_in[7];

    const int C  = in_sizes[3];            // 256
    const int F  = in_sizes[2] / C;        // 141
    const int N  = in_sizes[0] / F;        // 50000
    const int E  = in_sizes[1] / 2;        // 400000
    const int L  = in_sizes[5] / C;        // 4
    const int Kp = ((F + 31) / 32) * 32;   // 160

    const int* src = ei;
    const int* dst = ei + E;

    // -------- workspace carve-out --------
    char* ws = (char*)d_ws;
    size_t off = 0;
    auto take = [&](size_t bytes) { char* p = ws + off; off += align256(bytes); return p; };

    __bf16* xp     = (__bf16*)take((size_t)N * Kp * sizeof(__bf16));
    __bf16* WinB   = (__bf16*)take((size_t)C * Kp * sizeof(__bf16));
    __bf16* convWB = (__bf16*)take((size_t)L * C * C * sizeof(__bf16));
    float*  buf0   = (float*) take((size_t)N * C * sizeof(float));
    float*  buf1   = (float*) take((size_t)N * C * sizeof(float));
    float*  buf2   = (float*) take((size_t)N * C * sizeof(float));
    __bf16* hb     = (__bf16*)take((size_t)N * C * sizeof(__bf16));
    float*  dinv   = (float*) take((size_t)N * sizeof(float));
    float*  nrm    = (float*) take((size_t)E * sizeof(float));
    float*  stats  = (float*) take((size_t)2 * C * sizeof(float));
    (void)ws_size;

    const int TPB = 256;
    const int totNC = N * C;
    const int gNC = (totNC + TPB - 1) / TPB;

    // -------- bf16 conversions / padding --------
    k_pad_bf16<<<((size_t)N * Kp + TPB - 1) / TPB, TPB, 0, stream>>>(x, xp, N, F, Kp);
    k_pad_bf16<<<((size_t)C * Kp + TPB - 1) / TPB, TPB, 0, stream>>>(W_in, WinB, C, F, Kp);
    k_pad_bf16<<<((size_t)L * C * C + TPB - 1) / TPB, TPB, 0, stream>>>(conv_W, convWB, L * C, C, C);

    // -------- degree / symmetric normalization --------
    k_fill<<<(N + TPB - 1) / TPB, TPB, 0, stream>>>(dinv, 1.0f, N);  // self-loop
    k_deg_accum<<<(E + TPB - 1) / TPB, TPB, 0, stream>>>(dst, dinv, E);
    k_rsqrt_inplace<<<(N + TPB - 1) / TPB, TPB, 0, stream>>>(dinv, N);
    k_edge_norm<<<(E + TPB - 1) / TPB, TPB, 0, stream>>>(src, dst, dinv, nrm, E);

    // -------- input projection: h = relu(x @ W_in^T + b_in) --------
    k_gemm_bf16<<<N / 16, 128, 0, stream>>>(xp, WinB, buf0, Kp, N);
    k_bias_relu<<<gNC, TPB, 0, stream>>>(buf0, b_in, buf1, hb, totNC);

    float* h       = buf1;   // current activations (residual source)
    float* scratch = buf0;   // GEMM output m, then reused as h_out
    float* aggbuf  = buf2;   // aggregation buffer

    for (int i = 0; i < L; ++i) {
        // m = h @ conv_W[i]^T  (bf16 WMMA, f32 accumulate)
        k_gemm_bf16<<<N / 16, 128, 0, stream>>>(hb, convWB + (size_t)i * C * C,
                                                scratch, C, N);
        // agg = conv_b + self-loop term
        k_agg_init<<<gNC, TPB, 0, stream>>>(scratch, dinv, conv_b + (size_t)i * C,
                                            aggbuf, totNC);
        // agg[dst] += m[src] * norm
        k_scatter<<<E, C, 0, stream>>>(src, dst, nrm, scratch, aggbuf, E);
        // batchnorm statistics
        k_fill<<<(2 * C + TPB - 1) / TPB, TPB, 0, stream>>>(stats, 0.0f, 2 * C);
        k_stats_partial<<<512, C, 0, stream>>>(aggbuf, stats, stats + C, N);
        k_stats_final<<<1, C, 0, stream>>>(stats, stats + C, N);
        // h_out = relu(bn(agg)) + h   (writes over dead m buffer)
        k_bn_apply<<<gNC, TPB, 0, stream>>>(aggbuf, stats, stats + C,
                                            bn_gamma + (size_t)i * C,
                                            bn_beta + (size_t)i * C,
                                            h, scratch, hb, totNC);
        float* tmp = h; h = scratch; scratch = tmp;
    }

    hipMemcpyAsync(d_out, h, (size_t)N * C * sizeof(float),
                   hipMemcpyDeviceToDevice, stream);
}
